// LinearLoopLayer_21251498180716
// MI455X (gfx1250) — matrix operand order repaired and checked by reading, not compiled, not run on hardware
//
#include <hip/hip_runtime.h>
#include <hip/hip_bf16.h>

// ---------------------------------------------------------------------------
// out[b, o] = sum_k X[b,k] * W[o,k] + bias[o];  M = N = K = 4096, fp32 in/out.
//
// bf16x3 split-GEMM on V_WMMA_F32_16X16X32_BF16 (compute-bound).
//   Pass 1 (split_bf16): one-shot fp32 -> bf16 {hi,lo} planes into d_ws.
//   Pass 2 (gemm_bf16x3_async): 256x128 block tile, 8 waves * (64x64) each,
//           tiles DMA'd global->LDS via GLOBAL_LOAD_ASYNC_TO_LDS_B128
//           (ASYNCcnt), fragments via ds_load_b128, 48 WMMA per wave per
//           K=32 step at 0.67 ds_load/WMMA (LDS-bandwidth balanced).
// Fallback: fused convert-in-kernel variant if ws too small.
// ---------------------------------------------------------------------------

typedef __attribute__((ext_vector_type(16))) __bf16 v16bf;
typedef __attribute__((ext_vector_type(8)))  __bf16 v8bf;
typedef __attribute__((ext_vector_type(4)))  __bf16 v4bf;
typedef __attribute__((ext_vector_type(8)))  float  v8f;
typedef __attribute__((ext_vector_type(4)))  float  v4f;

#define TILE_K 32
#define LDSTR  40   // 32 + 8 bf16 pad: 80B row stride, conflict-free b128 reads

// async-kernel block geometry
#define AM 256      // block rows (M)
#define AN 128      // block cols (N)

// fallback block geometry
#define FM 128
#define FN 128

static __device__ __forceinline__ v16bf cat8(v8bf a, v8bf b) {
  return __builtin_shufflevector(a, b, 0,1,2,3,4,5,6,7,8,9,10,11,12,13,14,15);
}

#define WMMA_BF16(A, B, C) \
  __builtin_amdgcn_wmma_f32_16x16x32_bf16(false, (A), false, (B), (short)0, (C), false, false)

// ---------------------------------------------------------------------------
// Pass 1: split fp32 into bf16 hi + lo planes (hi = rn(bf16), lo = rn(x-hi)).
// ---------------------------------------------------------------------------
__global__ __launch_bounds__(256)
void split_bf16(const float* __restrict__ X, const float* __restrict__ W,
                unsigned short* __restrict__ Xh, unsigned short* __restrict__ Xl,
                unsigned short* __restrict__ Wh, unsigned short* __restrict__ Wl)
{
  const size_t NEL = (size_t)4096 * 4096;
  size_t gid = (size_t)blockIdx.x * 256 + threadIdx.x;
  size_t e = gid * 4;
  const float* src;
  unsigned short *dh, *dl;
  size_t off;
  if (e < NEL) { src = X; dh = Xh; dl = Xl; off = e; }
  else         { src = W; dh = Wh; dl = Wl; off = e - NEL; }
  v4f v = *(const v4f*)(src + off);
  v4bf h, l;
#pragma unroll
  for (int j = 0; j < 4; ++j) {
    __bf16 hv = (__bf16)v[j];
    h[j] = hv;
    l[j] = (__bf16)(v[j] - (float)hv);
  }
  *(v4bf*)(dh + off) = h;
  *(v4bf*)(dl + off) = l;
}

// ---------------------------------------------------------------------------
// Pass 2: bf16x3 WMMA GEMM, 64x64 per wave, async global->LDS tile DMA.
// ---------------------------------------------------------------------------
__global__ __launch_bounds__(256)
void gemm_bf16x3_async(const unsigned short* __restrict__ Xh,
                       const unsigned short* __restrict__ Xl,
                       const unsigned short* __restrict__ Wh,
                       const unsigned short* __restrict__ Wl,
                       const float* __restrict__ Bias,
                       float* __restrict__ Out,
                       int M, int N, int K)
{
  // [stage][plane hi/lo][row][col]: A = 80 KB, B = 40 KB -> 120 KB total.
  __shared__ __bf16 sA[2][2][AM][LDSTR];
  __shared__ __bf16 sB[2][2][AN][LDSTR];

  const int tid  = threadIdx.x;
  const int lane = tid & 31;
  const int wave = tid >> 5;
  const int wm   = wave & 3;     // 4 waves along M -> 64 rows each
  const int wn   = wave >> 2;    // 2 waves along N -> 64 cols each
  const int l16  = lane & 15;
  const int half = lane >> 4;
  const int m0 = blockIdx.y * AM;
  const int n0 = blockIdx.x * AN;

  v8f acc[4][4];
#pragma unroll
  for (int tm = 0; tm < 4; ++tm)
#pragma unroll
    for (int tn = 0; tn < 4; ++tn) {
      v8f z = {0.f,0.f,0.f,0.f,0.f,0.f,0.f,0.f};
      acc[tm][tn] = z;
    }

  // Async DMA one stage: A planes 2*1024 chunks + B planes 2*512 chunks
  // (16B each) = 3072 chunks, 12 per thread.
  auto issue_async = [&](int s, int kt) {
#pragma unroll
    for (int p = 0; p < 2; ++p) {
      const unsigned short* srcA = p ? Xl : Xh;
#pragma unroll
      for (int j = 0; j < 4; ++j) {
        int c   = tid + j * 256;     // 0..1023
        int row = c >> 2;
        int kc  = (c & 3) << 3;
        unsigned lds_addr = (unsigned)(size_t)&sA[s][p][row][kc];
        const unsigned short* g = srcA + (size_t)(m0 + row) * K + kt + kc;
        asm volatile("global_load_async_to_lds_b128 %0, %1, off"
                     :: "v"(lds_addr), "v"(g) : "memory");
      }
      const unsigned short* srcB = p ? Wl : Wh;
#pragma unroll
      for (int j = 0; j < 2; ++j) {
        int c   = tid + j * 256;     // 0..511
        int row = c >> 2;
        int kc  = (c & 3) << 3;
        unsigned lds_addr = (unsigned)(size_t)&sB[s][p][row][kc];
        const unsigned short* g = srcB + (size_t)(n0 + row) * K + kt + kc;
        asm volatile("global_load_async_to_lds_b128 %0, %1, off"
                     :: "v"(lds_addr), "v"(g) : "memory");
      }
    }
  };

  // One K=32 step: 4x4 tiles x 3 WMMAs (lo*hi, hi*lo, hi*hi) = 48 WMMA,
  // fed by 16 A + 16 B ds_load_b128.
  auto compute = [&](int s) {
    v16bf fah[4], fal[4];
#pragma unroll
    for (int tm = 0; tm < 4; ++tm) {
      int r  = wm * 64 + tm * 16 + l16;   // A row (M)
      int kb = half * 8;                   // ISA 16-bit A fragment layout
      fah[tm] = cat8(*(const v8bf*)&sA[s][0][r][kb],
                     *(const v8bf*)&sA[s][0][r][kb + 16]);
      fal[tm] = cat8(*(const v8bf*)&sA[s][1][r][kb],
                     *(const v8bf*)&sA[s][1][r][kb + 16]);
    }
#pragma unroll
    for (int tn = 0; tn < 4; ++tn) {
      int r  = wn * 64 + tn * 16 + l16;   // B column (N) = W row
      int kb = half * 8;                   // same k order as A fragment
      v16bf fbh = cat8(*(const v8bf*)&sB[s][0][r][kb],
                       *(const v8bf*)&sB[s][0][r][kb + 16]);
      v16bf fbl = cat8(*(const v8bf*)&sB[s][1][r][kb],
                       *(const v8bf*)&sB[s][1][r][kb + 16]);
#pragma unroll
      for (int tm = 0; tm < 4; ++tm) {
        acc[tm][tn] = WMMA_BF16(fal[tm], fbh, acc[tm][tn]);
        acc[tm][tn] = WMMA_BF16(fah[tm], fbl, acc[tm][tn]);
        acc[tm][tn] = WMMA_BF16(fah[tm], fbh, acc[tm][tn]);
      }
    }
  };

  const int NT = K / TILE_K;   // 128

  issue_async(0, 0);
  asm volatile("s_wait_asynccnt 0x0" ::: "memory");
  __syncthreads();

  for (int t = 0; t < NT; ++t) {
    int s = t & 1;
    if (t + 1 < NT) issue_async(s ^ 1, (t + 1) * TILE_K);  // DMA overlaps WMMA
    compute(s);
    asm volatile("s_wait_asynccnt 0x0" ::: "memory");
    __syncthreads();
  }

  // Epilogue: D layout -> VGPR v holds M = v + 8*half, N = lane%16.
#pragma unroll
  for (int tn = 0; tn < 4; ++tn) {
    int col = n0 + wn * 64 + tn * 16 + l16;
    float bv = Bias[col];
#pragma unroll
    for (int tm = 0; tm < 4; ++tm) {
      int rbase = m0 + wm * 64 + tm * 16 + half * 8;
#pragma unroll
      for (int v = 0; v < 8; ++v) {
        Out[(size_t)(rbase + v) * N + col] = acc[tm][tn][v] + bv;
      }
    }
  }
}

// ---------------------------------------------------------------------------
// Fallback: fused convert-in-kernel bf16x3.
// ---------------------------------------------------------------------------
__global__ __launch_bounds__(256)
void linear_bf16x3_fused(const float* __restrict__ X,
                         const float* __restrict__ W,
                         const float* __restrict__ Bias,
                         float* __restrict__ Out,
                         int M, int N, int K)
{
  __shared__ __bf16 sAh[2][FM][LDSTR];
  __shared__ __bf16 sAl[2][FM][LDSTR];
  __shared__ __bf16 sBh[2][FN][LDSTR];
  __shared__ __bf16 sBl[2][FN][LDSTR];

  const int tid  = threadIdx.x;
  const int lane = tid & 31;
  const int wave = tid >> 5;
  const int wm   = wave & 3;
  const int wn   = wave >> 2;
  const int l16  = lane & 15;
  const int half = lane >> 4;
  const int m0 = blockIdx.y * FM;
  const int n0 = blockIdx.x * FN;

  v8f acc[2][4];
#pragma unroll
  for (int tm = 0; tm < 2; ++tm)
#pragma unroll
    for (int tn = 0; tn < 4; ++tn) {
      v8f z = {0.f,0.f,0.f,0.f,0.f,0.f,0.f,0.f};
      acc[tm][tn] = z;
    }

  v4f rA[4], rB[4];

  auto gload = [&](int kt) {
#pragma unroll
    for (int i = 0; i < 4; ++i) {
      int idx = tid + i * 256;
      int r = idx >> 3;
      int c = (idx & 7) << 2;
      rA[i] = *(const v4f*)(X + (size_t)(m0 + r) * K + kt + c);
      rB[i] = *(const v4f*)(W + (size_t)(n0 + r) * K + kt + c);
    }
  };

  auto stlds = [&](int s) {
#pragma unroll
    for (int i = 0; i < 4; ++i) {
      int idx = tid + i * 256;
      int r = idx >> 3;
      int c = (idx & 7) << 2;
      v4bf ah, al, bh, bl;
#pragma unroll
      for (int j = 0; j < 4; ++j) {
        float a  = rA[i][j];
        __bf16 h = (__bf16)a;
        ah[j] = h;
        al[j] = (__bf16)(a - (float)h);
        float b  = rB[i][j];
        __bf16 g = (__bf16)b;
        bh[j] = g;
        bl[j] = (__bf16)(b - (float)g);
      }
      *(v4bf*)&sAh[s][r][c] = ah;
      *(v4bf*)&sAl[s][r][c] = al;
      *(v4bf*)&sBh[s][r][c] = bh;
      *(v4bf*)&sBl[s][r][c] = bl;
    }
  };

  auto compute = [&](int s) {
    v16bf fah[2], fal[2];
#pragma unroll
    for (int tm = 0; tm < 2; ++tm) {
      int r  = wm * 32 + tm * 16 + l16;
      int kb = half * 8;
      fah[tm] = cat8(*(const v8bf*)&sAh[s][r][kb],
                     *(const v8bf*)&sAh[s][r][kb + 16]);
      fal[tm] = cat8(*(const v8bf*)&sAl[s][r][kb],
                     *(const v8bf*)&sAl[s][r][kb + 16]);
    }
#pragma unroll
    for (int tn = 0; tn < 4; ++tn) {
      int r  = wn * 64 + tn * 16 + l16;
      int kb = half * 8;
      v16bf fbh = cat8(*(const v8bf*)&sBh[s][r][kb],
                       *(const v8bf*)&sBh[s][r][kb + 16]);
      v16bf fbl = cat8(*(const v8bf*)&sBl[s][r][kb],
                       *(const v8bf*)&sBl[s][r][kb + 16]);
#pragma unroll
      for (int tm = 0; tm < 2; ++tm) {
        acc[tm][tn] = WMMA_BF16(fal[tm], fbh, acc[tm][tn]);
        acc[tm][tn] = WMMA_BF16(fah[tm], fbl, acc[tm][tn]);
        acc[tm][tn] = WMMA_BF16(fah[tm], fbh, acc[tm][tn]);
      }
    }
  };

  const int NT = K / TILE_K;

  gload(0);
  stlds(0);
  __syncthreads();

  for (int t = 0; t < NT; ++t) {
    int s = t & 1;
    if (t + 1 < NT) gload((t + 1) * TILE_K);
    compute(s);
    if (t + 1 < NT) stlds(s ^ 1);
    __syncthreads();
  }

#pragma unroll
  for (int tn = 0; tn < 4; ++tn) {
    int col = n0 + wn * 64 + tn * 16 + l16;
    float bv = Bias[col];
#pragma unroll
    for (int tm = 0; tm < 2; ++tm) {
      int rbase = m0 + wm * 32 + tm * 16 + half * 8;
#pragma unroll
      for (int v = 0; v < 8; ++v) {
        Out[(size_t)(rbase + v) * N + col] = acc[tm][tn][v] + bv;
      }
    }
  }
}

extern "C" void kernel_launch(void* const* d_in, const int* in_sizes, int n_in,
                              void* d_out, int out_size, void* d_ws, size_t ws_size,
                              hipStream_t stream) {
  (void)in_sizes; (void)n_in; (void)out_size;
  const float* X    = (const float*)d_in[0];   // [4096, 4096]
  const float* W    = (const float*)d_in[1];   // [4096, 4096]
  const float* Bias = (const float*)d_in[2];   // [4096]
  float* Out        = (float*)d_out;           // [4096, 4096]

  const int M = 4096, N = 4096, K = 4096;
  const size_t NEL = (size_t)M * K;
  const size_t need = 4 * NEL * sizeof(unsigned short);   // 128 MB of planes

  dim3 block(256);

  if (ws_size >= need) {
    unsigned short* Xh = (unsigned short*)d_ws;
    unsigned short* Xl = Xh + NEL;
    unsigned short* Wh = Xl + NEL;
    unsigned short* Wl = Wh + NEL;
    split_bf16<<<dim3(32768), block, 0, stream>>>(X, W, Xh, Xl, Wh, Wl);
    dim3 grid(N / AN, M / AM);                 // 32 x 16 blocks
    gemm_bf16x3_async<<<grid, block, 0, stream>>>(Xh, Xl, Wh, Wl, Bias, Out, M, N, K);
  } else {
    dim3 grid(N / FN, M / FM);                 // 32 x 32 blocks
    linear_bf16x3_fused<<<grid, block, 0, stream>>>(X, W, Bias, Out, M, N, K);
  }
}
